// Attention_62826781606402
// MI455X (gfx1250) — matrix operand order repaired and checked by reading, not compiled, not run on hardware
//
#include <hip/hip_runtime.h>

typedef __attribute__((ext_vector_type(16))) __bf16 v16bf;
typedef __attribute__((ext_vector_type(8)))  float v8f;
typedef __attribute__((ext_vector_type(8)))  unsigned short u16x8;
typedef __attribute__((ext_vector_type(4)))  float f32x4;

union Frag {
  v16bf v;
  u16x8 h[2];
  unsigned short u[16];
};

__device__ __forceinline__ unsigned short f2bf(float f) {
  unsigned int u = __float_as_uint(f);
  u += 0x7FFFu + ((u >> 16) & 1u);   // round-to-nearest-even
  return (unsigned short)(u >> 16);
}

#define BATCH 8
#define SEQ   2048
#define CH    1024
#define MTOT  (BATCH * SEQ)

// ---------------------------------------------------------------------------
// Kernel 0: one-time fp32 -> bf16 conversion (8 elements / thread, 128-bit IO).
// ---------------------------------------------------------------------------
__global__ __launch_bounds__(256)
void cvt_bf16_kernel(const float* __restrict__ src,
                     unsigned short* __restrict__ dst, int n8)
{
  const int i = blockIdx.x * 256 + threadIdx.x;
  if (i >= n8) return;
  const float* p = src + (size_t)i * 8;
  f32x4 a = *(const f32x4*)(p);
  f32x4 b = *(const f32x4*)(p + 4);
  u16x8 r;
  for (int j = 0; j < 4; ++j) { r[j] = f2bf(a[j]); r[4 + j] = f2bf(b[j]); }
  *(u16x8*)(dst + (size_t)i * 8) = r;
}

// ---------------------------------------------------------------------------
// Kernel 1: q = x@Wq^T + bq (likewise k, v), all operands pre-converted bf16.
// One wave -> 32x64 output tile (2 M-subtiles x 4 N-tiles, B-frags reused).
// z==0 -> q (bf16 row-major), z==1 -> k (bf16 row-major), z==2 -> v transposed
// [B, C, T] so that P@V B-fragments are contiguous.
// ---------------------------------------------------------------------------
__global__ __launch_bounds__(256)
void qkv_proj_kernel(const unsigned short* __restrict__ xb,
                     const unsigned short* __restrict__ Wqb,
                     const unsigned short* __restrict__ Wkb,
                     const unsigned short* __restrict__ Wvb,
                     const float* __restrict__ bq,
                     const float* __restrict__ bk,
                     const float* __restrict__ bv,
                     unsigned short* __restrict__ qbuf,
                     unsigned short* __restrict__ kbuf,
                     unsigned short* __restrict__ vT)
{
  const int lane = threadIdx.x & 31;
  const int w    = threadIdx.x >> 5;
  const bool lo  = lane < 16;
  const int m0 = blockIdx.x * 32;
  const int n0 = blockIdx.y * 512 + w * 64;
  const int z  = blockIdx.z;

  const unsigned short* W = (z == 0) ? Wqb : (z == 1) ? Wkb : Wvb;
  const float* bias       = (z == 0) ? bq  : (z == 1) ? bk  : bv;

  v8f acc[2][4];
  for (int ms = 0; ms < 2; ++ms)
    for (int t = 0; t < 4; ++t)
      for (int i = 0; i < 8; ++i) acc[ms][t][i] = 0.0f;

  // A rows: subtile 0 row m0+lane%16, subtile 1 is +16 rows.
  const unsigned short* xrow0 = xb + (size_t)(m0 + (lane & 15)) * CH;
  const unsigned short* xrow1 = xrow0 + (size_t)16 * CH;
  // B rows (one per N-tile): W row n = col index (x@W^T -> contiguous K).
  const unsigned short* wrow[4];
  for (int t = 0; t < 4; ++t)
    wrow[t] = W + (size_t)(n0 + t * 16 + (lane & 15)) * CH;

  const int aoff = lo ? 0 : 8;    // A: lo lanes K 0..7/16..23, hi lanes +8
  const int boff = lo ? 0 : 8;    // B: lo lanes K 0..7/16..23, hi lanes +8

  for (int k = 0; k < CH; k += 32) {
    Frag af0, af1;
    af0.h[0] = *(const u16x8*)(xrow0 + k + aoff);
    af0.h[1] = *(const u16x8*)(xrow0 + k + aoff + 16);
    af1.h[0] = *(const u16x8*)(xrow1 + k + aoff);
    af1.h[1] = *(const u16x8*)(xrow1 + k + aoff + 16);
    for (int t = 0; t < 4; ++t) {
      Frag bfr;
      bfr.h[0] = *(const u16x8*)(wrow[t] + k + boff);
      bfr.h[1] = *(const u16x8*)(wrow[t] + k + boff + 16);
      acc[0][t] = __builtin_amdgcn_wmma_f32_16x16x32_bf16(
          false, af0.v, false, bfr.v, (short)0, acc[0][t], false, false);
      acc[1][t] = __builtin_amdgcn_wmma_f32_16x16x32_bf16(
          false, af1.v, false, bfr.v, (short)0, acc[1][t], false, false);
    }
  }

  // bias + store (C/D layout: row = j + 8*(lane>=16), col = n0+t*16+lane%16)
  for (int ms = 0; ms < 2; ++ms) {
    const int m0s = m0 + ms * 16;
    for (int t = 0; t < 4; ++t) {
      const int n = n0 + t * 16 + (lane & 15);
      const float bb = bias[n];
      if (z < 2) {
        unsigned short* out = (z == 0) ? qbuf : kbuf;
        for (int j = 0; j < 8; ++j) {
          const int m = m0s + j + (lo ? 0 : 8);
          out[(size_t)m * CH + n] = f2bf(acc[ms][t][j] + bb);
        }
      } else {
        const int batch = m0s >> 11;                    // m0s / SEQ
        const int tr = (m0s & (SEQ - 1)) + (lo ? 0 : 8);
        Frag pk;
        for (int j = 0; j < 8; ++j) pk.u[j] = f2bf(acc[ms][t][j] + bb);
        *(u16x8*)&vT[((size_t)batch * CH + n) * SEQ + tr] = pk.h[0];
      }
    }
  }
}

// ---------------------------------------------------------------------------
// Kernel 2: flash attention. One 512-thread workgroup per (batch, 16-query
// tile). Key blocks of 256: wave w owns 16-key slice of S and 64-col slice of O.
// ---------------------------------------------------------------------------
__global__ __launch_bounds__(512)
void flash_attn_kernel(const unsigned short* __restrict__ qb,
                       const unsigned short* __restrict__ kb,
                       const unsigned short* __restrict__ vT,
                       float* __restrict__ out)
{
  __shared__ unsigned short Qs[16 * CH];       // 32 KB: staged Q tile (bf16)
  __shared__ unsigned short Pb[16 * 256];      // 8 KB: exp'd probabilities
  __shared__ float red[16][16];                // per-wave row stats
  __shared__ float m_run[16], l_run[16], m_new_s[16], fac_s[16];

  const int lane = threadIdx.x & 31;
  const int w    = threadIdx.x >> 5;
  const bool lo  = lane < 16;
  const int qt = blockIdx.x;
  const int bz = blockIdx.y;
  const int qrow0 = qt * 16;

  // stage Q tile into LDS (16 x 1024 bf16) with 128-bit transfers
  {
    const int tid = threadIdx.x;
    for (int i = 0; i < 4; ++i) {
      const int idx = tid + i * 512;           // ushort8 chunk id (0..2047)
      const int row = idx >> 7;
      const int col = (idx & 127) * 8;
      *(u16x8*)&Qs[row * CH + col] =
          *(const u16x8*)&qb[((size_t)(bz * SEQ + qrow0 + row)) * CH + col];
    }
    if (w == 0 && lane < 16) { m_run[lane] = -1e30f; l_run[lane] = 0.0f; }
  }
  __syncthreads();

  v8f o[4];
  for (int t = 0; t < 4; ++t)
    for (int i = 0; i < 8; ++i) o[t][i] = 0.0f;

  const int nblk = (qt >> 4) + 1;              // ceil((qt+1)/16)
  const float scale = 0.03125f;                // 1/sqrt(1024)
  const int qmax = qrow0 + 15;

  for (int blk = 0; blk < nblk; ++blk) {
    const int kb0 = blk * 256;
    const int ks  = kb0 + w * 16;
    // causal limit within this key block, rounded up to WMMA K granularity
    const int kklim = min(256, ((qmax - kb0) | 31) + 1);

    // ---- Phase A: S = scale * Q K^T for this wave's 16-key slice ----
    v8f s;
    if (ks <= qmax) {
      for (int i = 0; i < 8; ++i) s[i] = 0.0f;
      const int key = ks + (lane & 15);
      const unsigned short* krow = kb + ((size_t)(bz * SEQ + key)) * CH;
      const int arow = lane & 15;
      for (int kc = 0; kc < CH; kc += 32) {
        Frag af, bfr;
        const unsigned short* qp = &Qs[arow * CH + kc + (lo ? 0 : 8)];
        af.h[0] = *(const u16x8*)(qp);
        af.h[1] = *(const u16x8*)(qp + 16);
        const unsigned short* kp = krow + kc + (lo ? 0 : 8);
        bfr.h[0] = *(const u16x8*)(kp);
        bfr.h[1] = *(const u16x8*)(kp + 16);
        s = __builtin_amdgcn_wmma_f32_16x16x32_bf16(
            false, af.v, false, bfr.v, (short)0, s, false, false);
      }
      for (int j = 0; j < 8; ++j) {            // scale + causal mask
        const int row = qrow0 + j + (lo ? 0 : 8);
        s[j] = (key <= row) ? s[j] * scale : -1e30f;
      }
    } else {
      for (int i = 0; i < 8; ++i) s[i] = -1e30f;
    }

    // local row-max, cross-wave reduce via LDS
    for (int j = 0; j < 8; ++j) {
      float v = s[j];
      v = fmaxf(v, __shfl_xor(v, 8, 32));
      v = fmaxf(v, __shfl_xor(v, 4, 32));
      v = fmaxf(v, __shfl_xor(v, 2, 32));
      v = fmaxf(v, __shfl_xor(v, 1, 32));
      if (lane == 0)  red[w][j]     = v;
      if (lane == 16) red[w][j + 8] = v;
    }
    __syncthreads();
    if (w == 0 && lane < 16) {
      float mb = red[0][lane];
      for (int ww = 1; ww < 16; ++ww) mb = fmaxf(mb, red[ww][lane]);
      const float mo = m_run[lane];
      const float mn = fmaxf(mo, mb);
      m_run[lane]   = mn;
      m_new_s[lane] = mn;
      fac_s[lane]   = __expf(mo - mn);
    }
    __syncthreads();

    // ---- Phase B: rescale O, P = exp(s - m_new), row sums ----
    for (int j = 0; j < 8; ++j) {
      const float fc = lo ? fac_s[j] : fac_s[j + 8];
      for (int t = 0; t < 4; ++t) o[t][j] *= fc;
    }
    // only the [0, kklim) range of Pb is consumed by Phase C
    if (ks < kb0 + kklim) {
      for (int j = 0; j < 8; ++j) {
        const float mn = lo ? m_new_s[j] : m_new_s[j + 8];
        const float p = __expf(s[j] - mn);
        Pb[(j + (lo ? 0 : 8)) * 256 + w * 16 + (lane & 15)] = f2bf(p);
        float v = p;
        v += __shfl_xor(v, 8, 32);
        v += __shfl_xor(v, 4, 32);
        v += __shfl_xor(v, 2, 32);
        v += __shfl_xor(v, 1, 32);
        if (lane == 0)  red[w][j]     = v;
        if (lane == 16) red[w][j + 8] = v;
      }
    } else {
      // row-sum contribution is exactly zero for fully-masked slices
      for (int j = 0; j < 8; ++j) {
        if (lane == 0)  red[w][j]     = 0.0f;
        if (lane == 16) red[w][j + 8] = 0.0f;
      }
    }
    __syncthreads();
    if (w == 0 && lane < 16) {
      float sb = 0.0f;
      for (int ww = 0; ww < 16; ++ww) sb += red[ww][lane];
      l_run[lane] = l_run[lane] * fac_s[lane] + sb;
    }

    // ---- Phase C: O(16x64 slice) += P(16x0..kklim) @ V(kklim x 64 slice) ----
    {
      const int arow = lane & 15;
      for (int kk = 0; kk < kklim; kk += 32) {
        Frag af;
        const unsigned short* pp = &Pb[arow * 256 + kk + (lo ? 0 : 8)];
        af.h[0] = *(const u16x8*)(pp);
        af.h[1] = *(const u16x8*)(pp + 16);
        const int kbase = kb0 + kk + (lo ? 0 : 8);
        for (int t = 0; t < 4; ++t) {
          const int cw = w * 64 + t * 16 + (lane & 15);
          const unsigned short* vp =
              vT + ((size_t)(bz * CH + cw)) * SEQ + kbase;
          Frag bfr;
          bfr.h[0] = *(const u16x8*)(vp);
          bfr.h[1] = *(const u16x8*)(vp + 16);
          o[t] = __builtin_amdgcn_wmma_f32_16x16x32_bf16(
              false, af.v, false, bfr.v, (short)0, o[t], false, false);
        }
      }
    }
    __syncthreads();   // Pb / stats safe to overwrite next block
  }

  // ---- finalize: O /= l_run, fp32 store ----
  for (int j = 0; j < 8; ++j) {
    const float inv = 1.0f / (lo ? l_run[j] : l_run[j + 8]);
    const int row = qrow0 + j + (lo ? 0 : 8);
    float* orow = out + ((size_t)(bz * SEQ + row)) * CH;
    for (int t = 0; t < 4; ++t)
      orow[w * 64 + t * 16 + (lane & 15)] = o[t][j] * inv;
  }
}

// ---------------------------------------------------------------------------
extern "C" void kernel_launch(void* const* d_in, const int* in_sizes, int n_in,
                              void* d_out, int out_size, void* d_ws, size_t ws_size,
                              hipStream_t stream) {
  const float* x  = (const float*)d_in[0];
  const float* Wq = (const float*)d_in[1];
  const float* bq = (const float*)d_in[2];
  const float* Wk = (const float*)d_in[3];
  const float* bk = (const float*)d_in[4];
  const float* Wv = (const float*)d_in[5];
  const float* bv = (const float*)d_in[6];
  float* out = (float*)d_out;

  const size_t elems = (size_t)BATCH * SEQ * CH;   // 16.8M
  const size_t welems = (size_t)CH * CH;           // 1.05M
  unsigned short* qb  = (unsigned short*)d_ws;     // 32 MB bf16 q  [B,T,C]
  unsigned short* kb  = qb + elems;                // 32 MB bf16 k  [B,T,C]
  unsigned short* vT  = kb + elems;                // 32 MB bf16 v  [B,C,T]
  unsigned short* xbb = vT + elems;                // 32 MB bf16 x  [B,T,C]
  unsigned short* wqb = xbb + elems;               //  2 MB bf16 Wq
  unsigned short* wkb = wqb + welems;              //  2 MB bf16 Wk
  unsigned short* wvb = wkb + welems;              //  2 MB bf16 Wv

  // one-time bf16 conversion of x and weights
  {
    const int nx8 = (int)(elems / 8);
    cvt_bf16_kernel<<<(nx8 + 255) / 256, 256, 0, stream>>>(x, xbb, nx8);
    const int nw8 = (int)(welems / 8);
    cvt_bf16_kernel<<<(nw8 + 255) / 256, 256, 0, stream>>>(Wq, wqb, nw8);
    cvt_bf16_kernel<<<(nw8 + 255) / 256, 256, 0, stream>>>(Wk, wkb, nw8);
    cvt_bf16_kernel<<<(nw8 + 255) / 256, 256, 0, stream>>>(Wv, wvb, nw8);
  }

  dim3 g1(MTOT / 32, 2, 3);   // 32-row tiles x N-half x {q,k,v}
  qkv_proj_kernel<<<g1, 256, 0, stream>>>(xbb, wqb, wkb, wvb, bq, bk, bv,
                                          qb, kb, vT);

  dim3 g2(SEQ / 16, BATCH);   // query tile x batch
  flash_attn_kernel<<<g2, 512, 0, stream>>>(qb, kb, vT, out);
}
